// BidirectionalMamba2_53128745452286
// MI455X (gfx1250) — compile-verified, matrix operand order checked by reading, not run on hardware
//
#include <hip/hip_runtime.h>
#include <hip/hip_bf16.h>

// ---------------------------------------------------------------------------
// Bidirectional Mamba2 for MI455X (gfx1250): bf16 WMMA everywhere, fused
// elementwise stages, streaming inter-chunk scan. Async-to-LDS tile staging
// when the toolchain exposes the gfx1250 async builtins.
// ---------------------------------------------------------------------------

typedef __attribute__((ext_vector_type(16))) __bf16 v16bf;
typedef __attribute__((ext_vector_type(8)))  __bf16 v8bf;
typedef __attribute__((ext_vector_type(8)))  float  v8f;
typedef int v4i __attribute__((vector_size(16)));   // GCC-vector int4 for async builtin

#define NB    4
#define LSEQ  8192
#define MROWS (NB * LSEQ)   // 32768

#if defined(__HIP_DEVICE_COMPILE__) && __has_builtin(__builtin_amdgcn_global_load_async_to_lds_b128)
#define USE_ASYNC_LDS 1
#endif

#ifdef USE_ASYNC_LDS
__device__ __forceinline__ void async_cp16(const __bf16* g, __bf16* l) {
  // signature (probe-derived): (v4i addrspace(1)* src, v4i addrspace(3)* dst, imm off, imm cpol)
  __builtin_amdgcn_global_load_async_to_lds_b128(
      (__attribute__((address_space(1))) v4i*)(g),
      (__attribute__((address_space(3))) v4i*)(l), 0, 0);
}
__device__ __forceinline__ void async_wait() {
#if __has_builtin(__builtin_amdgcn_s_wait_asynccnt)
  __builtin_amdgcn_s_wait_asynccnt(0);
#else
  asm volatile("s_wait_asynccnt 0x0" ::: "memory");
#endif
}
#endif

__device__ __forceinline__ v8f wmma_bf16(v16bf a, v16bf b, v8f c) {
  return __builtin_amdgcn_wmma_f32_16x16x32_bf16(false, a, false, b,
                                                 (short)0, c, false, false);
}

__device__ __forceinline__ v8f vzero() {
  v8f z = {0.f, 0.f, 0.f, 0.f, 0.f, 0.f, 0.f, 0.f};
  return z;
}

// Load a 16x32 bf16 operand fragment from LDS (stride in elements, multiple
// of 8 so the 16B vector loads are aligned). Layout per CDNA5 ISA: lane
// (L&15) = row, half-wave selects K-base 0/8; halves 0..7 = K[kb..kb+7],
// halves 8..15 = K[kb+16..kb+23].
__device__ __forceinline__ v16bf ldfrag(const __bf16* p, int stride) {
  int lane = threadIdx.x & 31;
  const __bf16* r = p + (lane & 15) * stride + ((lane >> 4) << 3);
  union { v16bf v; v8bf h[2]; } u;
  u.h[0] = *(const v8bf*)(r);
  u.h[1] = *(const v8bf*)(r + 16);
  return u.v;
}

// ---------------------------------------------------------------------------
// fp32 -> bf16 convert
__global__ void k_cvt(const float* __restrict__ s, __bf16* __restrict__ d, int n) {
  int i = blockIdx.x * 256 + threadIdx.x;
  if (i < n) d[i] = (__bf16)s[i];
}

// ---------------------------------------------------------------------------
// x (b, 256, L) fp32  ->  xt (b, L, 256) bf16   (tiled LDS transpose)
__global__ void k_transpose(const float* __restrict__ x, __bf16* __restrict__ xt) {
  __shared__ float t[32][33];
  int b  = blockIdx.z;
  int l0 = blockIdx.x * 32, c0 = blockIdx.y * 32;
  for (int i = threadIdx.y; i < 32; i += 8)
    t[i][threadIdx.x] = x[((size_t)b * 256 + c0 + i) * LSEQ + l0 + threadIdx.x];
  __syncthreads();
  for (int i = threadIdx.y; i < 32; i += 8)
    xt[((size_t)b * LSEQ + l0 + i) * 256 + c0 + threadIdx.x] = (__bf16)t[threadIdx.x][i];
}

// ---------------------------------------------------------------------------
// Generic bf16 WMMA GEMM: out = A(M x K) * Bw(N x K)^T
// grid = (M/128, N/128), block = 256 (8 waves, 4x2 wave grid, 32x64 per wave)
// EPI 0: store bf16 (fc_in -> u)
// EPI 1: store bf16 (in-proj -> zx), A rows gathered reversed when rev=1
// EPI 2: store/add fp32 into ysum (out-proj), row scatter reversed when rev=1
// EPI 3: store fp32 transposed into d_out (b, 256, L)  (fc_out)
template <int EPI>
__global__ void k_gemm(const __bf16* __restrict__ A, const __bf16* __restrict__ Bw,
                       void* __restrict__ outp, int K, int Nld, int rev, int add) {
  __shared__ __align__(16) __bf16 sA[128 * 40];
  __shared__ __align__(16) __bf16 sB[128 * 40];
  int tid  = threadIdx.x;
  int wid  = tid >> 5, lane = tid & 31;
  int wm   = wid & 3,  wn   = wid >> 2;
  int mB   = blockIdx.x * 128, nB = blockIdx.y * 128;

  v8f acc[2][4];
#pragma unroll
  for (int i = 0; i < 2; ++i)
#pragma unroll
    for (int j = 0; j < 4; ++j) acc[i][j] = vzero();

  int r    = tid >> 1;
  int cseg = (tid & 1) << 4;
  int gm   = mB + r;
  int grow = gm;
  if (EPI == 1 && rev) {
    int bb = gm >> 13, l = gm & (LSEQ - 1);
    grow = (bb << 13) + (LSEQ - 1 - l);
  }
  const __bf16* arow = A + (size_t)grow * K + cseg;
  const __bf16* brow = Bw + (size_t)(nB + r) * K + cseg;
  __bf16* la = sA + r * 40 + cseg;
  __bf16* lb = sB + r * 40 + cseg;

  for (int k0 = 0; k0 < K; k0 += 32) {
#ifdef USE_ASYNC_LDS
    async_cp16(arow + k0,     la);
    async_cp16(arow + k0 + 8, la + 8);
    async_cp16(brow + k0,     lb);
    async_cp16(brow + k0 + 8, lb + 8);
    if (k0 + 32 < K) {
      __builtin_prefetch(arow + k0 + 32, 0, 0);  // global_prefetch_b8
      __builtin_prefetch(brow + k0 + 32, 0, 0);
    }
    async_wait();
#else
    *(v8bf*)(la)     = *(const v8bf*)(arow + k0);
    *(v8bf*)(la + 8) = *(const v8bf*)(arow + k0 + 8);
    *(v8bf*)(lb)     = *(const v8bf*)(brow + k0);
    *(v8bf*)(lb + 8) = *(const v8bf*)(brow + k0 + 8);
    if (k0 + 32 < K) {
      __builtin_prefetch(arow + k0 + 32, 0, 0);
      __builtin_prefetch(brow + k0 + 32, 0, 0);
    }
#endif
    __syncthreads();
    v16bf af[2], bfr[4];
#pragma unroll
    for (int i = 0; i < 2; ++i) af[i]  = ldfrag(sA + (wm * 32 + i * 16) * 40, 40);
#pragma unroll
    for (int j = 0; j < 4; ++j) bfr[j] = ldfrag(sB + (wn * 64 + j * 16) * 40, 40);
#pragma unroll
    for (int i = 0; i < 2; ++i)
#pragma unroll
      for (int j = 0; j < 4; ++j) acc[i][j] = wmma_bf16(af[i], bfr[j], acc[i][j]);
    __syncthreads();
  }

#pragma unroll
  for (int i = 0; i < 2; ++i)
#pragma unroll
    for (int j = 0; j < 4; ++j) {
      int n = nB + wn * 64 + j * 16 + (lane & 15);
#pragma unroll
      for (int rr = 0; rr < 8; ++rr) {
        int m  = mB + wm * 32 + i * 16 + rr + ((lane >> 4) << 3);
        float v = acc[i][j][rr];
        if constexpr (EPI == 0 || EPI == 1) {
          ((__bf16*)outp)[(size_t)m * Nld + n] = (__bf16)v;
        } else if constexpr (EPI == 2) {
          int bb = m >> 13, l = m & (LSEQ - 1);
          int om = rev ? (bb << 13) + (LSEQ - 1 - l) : m;
          float* o = (float*)outp + (size_t)om * 256 + n;
          if (add) *o += v; else *o = v;
        } else {  // EPI == 3: transposed fp32 store
          int bb = m >> 13, l = m & (LSEQ - 1);
          ((float*)outp)[((size_t)bb * 256 + n) * LSEQ + l] = v;
        }
      }
    }
}

// ---------------------------------------------------------------------------
// dt = softplus(u . Win[1280+h] + dtbias[h])   (8 skinny columns, pure VALU)
__global__ void k_dt(const __bf16* __restrict__ u, const __bf16* __restrict__ Win,
                     const float* __restrict__ dtbias, float* __restrict__ dtout, int rev) {
  int e = blockIdx.x * 256 + threadIdx.x;    // MROWS*8
  int m = e >> 3, h = e & 7;
  int b = m >> 13, l = m & (LSEQ - 1);
  int gr = rev ? (b << 13) + (LSEQ - 1 - l) : m;
  const __bf16* a = u + (size_t)gr * 256;
  const __bf16* w = Win + (size_t)(1280 + h) * 256;
  float s = 0.f;
  for (int k = 0; k < 256; k += 8) {
    v8bf av = *(const v8bf*)(a + k);
    v8bf wv = *(const v8bf*)(w + k);
#pragma unroll
    for (int q = 0; q < 8; ++q) s += (float)av[q] * (float)wv[q];
  }
  s += dtbias[h];
  float sp = (s > 20.f) ? s : log1pf(__expf(s));
  dtout[(size_t)m * 8 + h] = sp;
}

// ---------------------------------------------------------------------------
// Causal depthwise conv (width 4) + bias + SiLU over xBC (cols 512..1279 of zx)
__global__ void k_conv(const __bf16* __restrict__ zx, const float* __restrict__ cw,
                       const float* __restrict__ cb, __bf16* __restrict__ xh,
                       __bf16* __restrict__ Bm, __bf16* __restrict__ Cm) {
  int e = blockIdx.x * 256 + threadIdx.x;    // MROWS*768
  int m = e / 768, ch = e - (e / 768) * 768;
  int l = m & (LSEQ - 1);
  float v = cb[ch];
#pragma unroll
  for (int k = 0; k < 4; ++k) {
    int ll = l - 3 + k;
    if (ll >= 0)
      v += (float)zx[(size_t)(m - 3 + k) * 1280 + 512 + ch] * cw[ch * 4 + k];
  }
  float s = v * (1.f / (1.f + __expf(-v)));  // SiLU
  if (ch < 512)      xh[(size_t)m * 512 + ch]       = (__bf16)s;
  else if (ch < 640) Bm[(size_t)m * 128 + ch - 512] = (__bf16)s;
  else               Cm[(size_t)m * 128 + ch - 640] = (__bf16)s;
}

// ---------------------------------------------------------------------------
// SSD intra-chunk: one block per (b, chunk, head).
// Yd = (L ⊙ (C B^T)) @ (x·dt) + D·x,  states = (x·dt)^T @ (decay·B),
// chunk_decay = exp(Acum[63]).
__global__ void k_ssd1(const __bf16* __restrict__ Cm, const __bf16* __restrict__ Bm,
                       const __bf16* __restrict__ xh, const float* __restrict__ dt,
                       const float* __restrict__ Alog, const float* __restrict__ Dv,
                       float* __restrict__ y, float* __restrict__ st,
                       float* __restrict__ cdec) {
  __shared__ __align__(16) char sm[62976];
  __bf16* sC  = (__bf16*)sm;             // 64 x 136
  __bf16* sB  = (__bf16*)(sm + 17408);   // 64 x 136
  __bf16* sBT = (__bf16*)(sm + 34816);   // 128 x 72  (decay-scaled B^T)
  __bf16* sXT = (__bf16*)(sm + 53248);   // 64 x 72   ((x*dt)^T)
  float*  sAc = (float*)(sm + 62464);    // 64
  float*  sDc = (float*)(sm + 62720);    // 64
  __bf16* sG  = (__bf16*)sm;             // alias of sC after G phase, 64 x 72

  int id = blockIdx.x;
  int h = id & 7, c = (id >> 3) & 127, b = id >> 10;
  int m0 = b * LSEQ + c * 64;
  int tid = threadIdx.x, wid = tid >> 5, lane = tid & 31;

  for (int i = tid; i < 1024; i += 256) {
    int row = i >> 4, seg = (i & 15) << 3;
    *(v8bf*)(sC + row * 136 + seg) = *(const v8bf*)(Cm + (size_t)(m0 + row) * 128 + seg);
    *(v8bf*)(sB + row * 136 + seg) = *(const v8bf*)(Bm + (size_t)(m0 + row) * 128 + seg);
  }
  float Ah = -__expf(Alog[h]);
  if (tid < 64) sDc[tid] = dt[(size_t)(m0 + tid) * 8 + h] * Ah;
  for (int i = tid; i < 4096; i += 256) {
    int l = i & 63, p = i >> 6;
    float xv = (float)xh[(size_t)(m0 + l) * 512 + h * 64 + p];
    sXT[p * 72 + l] = (__bf16)(xv * dt[(size_t)(m0 + l) * 8 + h]);
  }
  __syncthreads();
  if (tid == 0) {
    float s = 0.f;
    for (int l = 0; l < 64; ++l) { s += sDc[l]; sAc[l] = s; }
  }
  __syncthreads();
  if (tid < 64) sDc[tid] = __expf(sAc[63] - sAc[tid]);   // decay_states
  __syncthreads();

  // --- G = C @ B^T, masked + scaled by L = exp(Acum[l]-Acum[s]), s<=l ---
  float gq[2][8];
#pragma unroll
  for (int it = 0; it < 2; ++it) {
    int t = wid * 2 + it, tm = t >> 2, tn = t & 3;
    v8f acc = vzero();
#pragma unroll
    for (int k0 = 0; k0 < 128; k0 += 32)
      acc = wmma_bf16(ldfrag(sC + tm * 16 * 136 + k0, 136),
                      ldfrag(sB + tn * 16 * 136 + k0, 136), acc);
    int sidx  = tn * 16 + (lane & 15);
    int lbase = tm * 16 + ((lane >> 4) << 3);
    float as = sAc[sidx];
    float la[8];
#pragma unroll
    for (int r = 0; r < 8; ++r) la[r] = sAc[lbase + r];   // unconditional LDS reads
#pragma unroll
    for (int r = 0; r < 8; ++r) {
      float ev = __expf(la[r] - as);                       // branchless mask
      gq[it][r] = (sidx <= lbase + r) ? acc[r] * ev : 0.f;
    }
  }
  __syncthreads();
#pragma unroll
  for (int it = 0; it < 2; ++it) {
    int t = wid * 2 + it, tm = t >> 2, tn = t & 3;
    int sidx  = tn * 16 + (lane & 15);
    int lbase = tm * 16 + ((lane >> 4) << 3);
#pragma unroll
    for (int r = 0; r < 8; ++r)
      sG[(lbase + r) * 72 + sidx] = (__bf16)gq[it][r];
  }
  for (int i = tid; i < 8192; i += 256) {
    int l = i & 63, n = i >> 6;
    sBT[n * 72 + l] = (__bf16)((float)sB[l * 136 + n] * sDc[l]);
  }
  __syncthreads();

  // --- Yd = G @ (x*dt)  (+ D * x skip) ---
  float Dh = Dv[h];
#pragma unroll
  for (int it = 0; it < 2; ++it) {
    int t = wid * 2 + it, tm = t >> 2, tn = t & 3;
    v8f acc = vzero();
#pragma unroll
    for (int k0 = 0; k0 < 64; k0 += 32)
      acc = wmma_bf16(ldfrag(sG + tm * 16 * 72 + k0, 72),
                      ldfrag(sXT + tn * 16 * 72 + k0, 72), acc);
    int p = tn * 16 + (lane & 15);
#pragma unroll
    for (int r = 0; r < 8; ++r) {
      int l = tm * 16 + r + ((lane >> 4) << 3);
      size_t o = (size_t)(m0 + l) * 512 + h * 64 + p;
      y[o] = acc[r] + Dh * (float)xh[o];
    }
  }

  // --- states[p,n] = sum_l (x*dt)[l,p] * decay[l] * B[l,n] ---
#pragma unroll
  for (int it = 0; it < 4; ++it) {
    int t = wid * 4 + it, tm = t >> 3, tn = t & 7;
    v8f acc = vzero();
#pragma unroll
    for (int k0 = 0; k0 < 64; k0 += 32)
      acc = wmma_bf16(ldfrag(sXT + tm * 16 * 72 + k0, 72),
                      ldfrag(sBT + tn * 16 * 72 + k0, 72), acc);
    int n = tn * 16 + (lane & 15);
#pragma unroll
    for (int r = 0; r < 8; ++r) {
      int p = tm * 16 + r + ((lane >> 4) << 3);
      st[(size_t)id * 8192 + p * 128 + n] = acc[r];
    }
  }
  if (tid == 0) cdec[id] = __expf(sAc[63]);
}

// ---------------------------------------------------------------------------
// Inter-chunk scan (in-place states -> prev): one thread per (b,h,p,n)
__global__ void k_scan(float* __restrict__ st, const float* __restrict__ cdec) {
  int e = blockIdx.x * 256 + threadIdx.x;   // 4*8*8192
  int bh = e >> 13;
  int pn = e & 8191;
  int b = bh >> 3, h = bh & 7;
  float carry = 0.f;
  for (int c = 0; c < 128; ++c) {
    size_t idx = ((size_t)((b * 128 + c) * 8 + h)) * 8192 + pn;
    float s = st[idx];
    st[idx] = carry;                       // prev (state before this chunk)
    carry = carry * cdec[(b * 128 + c) * 8 + h] + s;
  }
}

// ---------------------------------------------------------------------------
// SSD inter-chunk: Yo = exp(Acum[l]) * (C @ prev^T), accumulated into y
__global__ void k_ssd2(const __bf16* __restrict__ Cm, const float* __restrict__ st,
                       const float* __restrict__ dt, const float* __restrict__ Alog,
                       float* __restrict__ y) {
  __shared__ __align__(16) char sm[35328];
  __bf16* sC  = (__bf16*)sm;             // 64 x 136
  __bf16* sP  = (__bf16*)(sm + 17408);   // 64 x 136 (prev[p][n])
  float*  sAc = (float*)(sm + 34816);    // 64
  float*  sT  = (float*)(sm + 35072);    // 64
  int id = blockIdx.x;
  int h = id & 7, c = (id >> 3) & 127, b = id >> 10;
  int m0 = b * LSEQ + c * 64;
  int tid = threadIdx.x, wid = tid >> 5, lane = tid & 31;

  for (int i = tid; i < 1024; i += 256) {
    int row = i >> 4, seg = (i & 15) << 3;
    *(v8bf*)(sC + row * 136 + seg) = *(const v8bf*)(Cm + (size_t)(m0 + row) * 128 + seg);
  }
  for (int i = tid; i < 8192; i += 256) {
    int p = i >> 7, n = i & 127;
    sP[p * 136 + n] = (__bf16)st[(size_t)id * 8192 + i];
  }
  float Ah = -__expf(Alog[h]);
  if (tid < 64) sT[tid] = dt[(size_t)(m0 + tid) * 8 + h] * Ah;
  __syncthreads();
  if (tid == 0) {
    float s = 0.f;
    for (int l = 0; l < 64; ++l) { s += sT[l]; sAc[l] = s; }
  }
  __syncthreads();

#pragma unroll
  for (int it = 0; it < 2; ++it) {
    int t = wid * 2 + it, tm = t >> 2, tn = t & 3;
    v8f acc = vzero();
#pragma unroll
    for (int k0 = 0; k0 < 128; k0 += 32)
      acc = wmma_bf16(ldfrag(sC + tm * 16 * 136 + k0, 136),
                      ldfrag(sP + tn * 16 * 136 + k0, 136), acc);
    int p = tn * 16 + (lane & 15);
    int lbase = tm * 16 + ((lane >> 4) << 3);
    float la[8];
#pragma unroll
    for (int r = 0; r < 8; ++r) la[r] = sAc[lbase + r];
#pragma unroll
    for (int r = 0; r < 8; ++r) {
      size_t o = (size_t)(m0 + lbase + r) * 512 + h * 64 + p;
      y[o] += __expf(la[r]) * acc[r];
    }
  }
}

// ---------------------------------------------------------------------------
// Gate with silu(z) + RMSNorm over d_inner=512, emit bf16. One block per row.
__global__ void k_norm(const float* __restrict__ y, const __bf16* __restrict__ zx,
                       const float* __restrict__ normw, __bf16* __restrict__ yn) {
  __shared__ float red[256];
  int m = blockIdx.x, t = threadIdx.x;
  float g[2];
#pragma unroll
  for (int j = 0; j < 2; ++j) {
    int i = t + j * 256;
    float z = (float)zx[(size_t)m * 1280 + i];
    float zi = z * (1.f / (1.f + __expf(-z)));
    g[j] = y[(size_t)m * 512 + i] * zi;
  }
  red[t] = g[0] * g[0] + g[1] * g[1];
  __syncthreads();
  for (int s = 128; s > 0; s >>= 1) {
    if (t < s) red[t] += red[t + s];
    __syncthreads();
  }
  float r = rsqrtf(red[0] * (1.f / 512.f) + 1e-5f);
#pragma unroll
  for (int j = 0; j < 2; ++j) {
    int i = t + j * 256;
    yn[(size_t)m * 512 + i] = (__bf16)(g[j] * r * normw[i]);
  }
}

// ---------------------------------------------------------------------------
extern "C" void kernel_launch(void* const* d_in, const int* in_sizes, int n_in,
                              void* d_out, int out_size, void* d_ws, size_t ws_size,
                              hipStream_t stream) {
  (void)in_sizes; (void)n_in; (void)out_size; (void)ws_size;
  const float* x       = (const float*)d_in[0];
  const float* Wfc_in  = (const float*)d_in[1];
  const float* Wfc_out = (const float*)d_in[2];
  const float* Win[2]    = {(const float*)d_in[3],  (const float*)d_in[11]};
  const float* convw[2]  = {(const float*)d_in[4],  (const float*)d_in[12]};
  const float* convb[2]  = {(const float*)d_in[5],  (const float*)d_in[13]};
  const float* Alog[2]   = {(const float*)d_in[6],  (const float*)d_in[14]};
  const float* Dv[2]     = {(const float*)d_in[7],  (const float*)d_in[15]};
  const float* dtbias[2] = {(const float*)d_in[8],  (const float*)d_in[16]};
  const float* normw[2]  = {(const float*)d_in[9],  (const float*)d_in[17]};
  const float* Wout[2]   = {(const float*)d_in[10], (const float*)d_in[18]};

  char* wp = (char*)d_ws;
  auto take = [&](size_t n) { char* r = wp; wp += (n + 255) & ~(size_t)255; return r; };

  __bf16* wfcin   = (__bf16*)take((size_t)256 * 256 * 2);
  __bf16* wwin0   = (__bf16*)take((size_t)1288 * 256 * 2);
  __bf16* wwin1   = (__bf16*)take((size_t)1288 * 256 * 2);
  __bf16* wwout0  = (__bf16*)take((size_t)256 * 512 * 2);
  __bf16* wwout1  = (__bf16*)take((size_t)256 * 512 * 2);
  __bf16* wfcout  = (__bf16*)take((size_t)256 * 256 * 2);
  __bf16* xtb     = (__bf16*)take((size_t)MROWS * 256 * 2);
  __bf16* u       = (__bf16*)take((size_t)MROWS * 256 * 2);
  __bf16* zx      = (__bf16*)take((size_t)MROWS * 1280 * 2);
  float*  dtb     = (float*) take((size_t)MROWS * 8 * 4);
  __bf16* xh      = (__bf16*)take((size_t)MROWS * 512 * 2);
  __bf16* Bmb     = (__bf16*)take((size_t)MROWS * 128 * 2);
  __bf16* Cmb     = (__bf16*)take((size_t)MROWS * 128 * 2);
  float*  st      = (float*) take((size_t)4 * 128 * 8 * 8192 * 4);
  float*  cdec    = (float*) take((size_t)4096 * 4);
  float*  ybuf    = (float*) take((size_t)MROWS * 512 * 4);
  __bf16* yn      = (__bf16*)take((size_t)MROWS * 512 * 2);
  float*  ysum    = (float*) take((size_t)MROWS * 256 * 4);
  __bf16* ysb     = (__bf16*)take((size_t)MROWS * 256 * 2);
  __bf16* wwin[2]  = {wwin0, wwin1};
  __bf16* wwout[2] = {wwout0, wwout1};

  // Weight conversion to bf16
  k_cvt<<<(65536 + 255) / 256, 256, 0, stream>>>(Wfc_in, wfcin, 65536);
  k_cvt<<<(329728 + 255) / 256, 256, 0, stream>>>(Win[0], wwin0, 329728);
  k_cvt<<<(329728 + 255) / 256, 256, 0, stream>>>(Win[1], wwin1, 329728);
  k_cvt<<<(131072 + 255) / 256, 256, 0, stream>>>(Wout[0], wwout0, 131072);
  k_cvt<<<(131072 + 255) / 256, 256, 0, stream>>>(Wout[1], wwout1, 131072);
  k_cvt<<<(65536 + 255) / 256, 256, 0, stream>>>(Wfc_out, wfcout, 65536);

  // Transpose + fc_in GEMM
  k_transpose<<<dim3(LSEQ / 32, 8, NB), dim3(32, 8), 0, stream>>>(x, xtb);
  k_gemm<0><<<dim3(MROWS / 128, 2), 256, 0, stream>>>(xtb, wfcin, u, 256, 256, 0, 0);

  for (int d = 0; d < 2; ++d) {
    // in-proj (z + xBC; N=1280), dt columns separately
    k_gemm<1><<<dim3(MROWS / 128, 10), 256, 0, stream>>>(u, wwin[d], zx, 256, 1280, d, 0);
    k_dt<<<(MROWS * 8) / 256, 256, 0, stream>>>(u, wwin[d], dtbias[d], dtb, d);
    // causal conv + SiLU
    k_conv<<<(MROWS * 768) / 256, 256, 0, stream>>>(zx, convw[d], convb[d], xh, Bmb, Cmb);
    // chunked SSD
    k_ssd1<<<4096, 256, 0, stream>>>(Cmb, Bmb, xh, dtb, Alog[d], Dv[d], ybuf, st, cdec);
    k_scan<<<(4 * 8 * 8192) / 256, 256, 0, stream>>>(st, cdec);
    k_ssd2<<<4096, 256, 0, stream>>>(Cmb, st, dtb, Alog[d], ybuf);
    // gate + RMSNorm
    k_norm<<<MROWS, 256, 0, stream>>>(ybuf, zx, normw[d], yn);
    // out-proj, accumulate into ysum (reversed scatter for backward dir)
    k_gemm<2><<<dim3(MROWS / 128, 2), 256, 0, stream>>>(yn, wwout[d], ysum, 512, 256, d, d);
  }

  // fc_out + transpose back to (b, 256, L)
  k_cvt<<<(MROWS * 256) / 256, 256, 0, stream>>>(ysum, ysb, MROWS * 256);
  k_gemm<3><<<dim3(MROWS / 128, 2), 256, 0, stream>>>(ysb, wfcout, d_out, 256, 256, 0, 0);
}
